// Correlation_5205500362854
// MI455X (gfx1250) — matrix operand order repaired and checked by reading, not compiled, not run on hardware
//
#include <hip/hip_runtime.h>

typedef __attribute__((ext_vector_type(16))) _Float16 v16h;
typedef __attribute__((ext_vector_type(8)))  float    v8f;
typedef __attribute__((ext_vector_type(4)))  float    f32x4;

#define BB 4
#define HH 192
#define WW 640
#define CC 32
#define SS 32

__device__ __forceinline__ int iclamp(int v, int lo, int hi) {
    return v < lo ? lo : (v > hi ? hi : v);
}

__device__ __forceinline__ float bcast_f(float v, int srclane) {
    return __uint_as_float(__builtin_amdgcn_readlane(__float_as_uint(v), srclane));
}
__device__ __forceinline__ int bcast_i(int v, int srclane) {
    return __builtin_amdgcn_readlane(v, srclane);
}

// One wave = 16 output pixels (fixed b,h), all 32 steps.
// A operand = f16 y-tile (16x32), persistent across steps.
// B operand = f16 warped-x tile transposed (32x16), rebuilt per step.
// D = A*B: diagonal(j,j) = sum_c y[w_j,c] * warp(x)[w_j,c]  -> corr * 32.
// Per-step warp parameters are computed lane-parallel (lane == step) before
// the loop: one division chain total instead of 32, broadcast via readlane.
__global__ __launch_bounds__(256) void corr_wmma_kernel(
    const float* __restrict__ x, const float* __restrict__ y,
    const float* __restrict__ origin, const float* __restrict__ focal,
    const float* __restrict__ T12, float* __restrict__ out)
{
    const int lane = threadIdx.x & 31;
    const int wave = threadIdx.x >> 5;

    int tile = blockIdx.x;
    const int wt = tile % (WW / 128);
    tile /= (WW / 128);
    const int h = tile % HH;
    const int b = tile / HH;

    const int w0 = wt * 128 + wave * 16;
    const int n  = lane & 15;        // pixel within wave tile
    const int pw = w0 + n;           // global w of this lane's column

    // Per-batch scalars (uniform across the block).
    const float tz  = T12[b * 3 + 2];
    const float t0  = T12[b * 3 + 0];
    const float t1  = T12[b * 3 + 1];
    const float ox  = origin[b * 2 + 0];
    const float oy  = origin[b * 2 + 1];
    const float fcx = focal[b * 2 + 0];
    const float fcy = focal[b * 2 + 1];

    // ---- Lane-parallel per-step parameters: lane <-> step d ----
    // d=0: 1/0 = +inf, 1/(inf+tz) = +0 == reference's guarded D=0.
    const float dfl     = (float)lane;
    const float Dl      = 1.0f / (1.0f / dfl + tz);
    const float Dtzl    = Dl * tz;
    const float alpha_l = 1.0f - Dtzl;
    const float beta_l  = Dtzl * ox + Dl * fcx * t0;
    const float gamma_l = Dtzl * oy + Dl * fcy * t1;
    // Row sampling is uniform per (b,h,d): fold into the precompute.
    const float sy_l  = alpha_l * (float)h + gamma_l;
    const float y0f_l = floorf(sy_l);
    const float fy_l  = sy_l - y0f_l;
    const int   y0i_l = (int)y0f_l;
    const int   y1i_l = y0i_l + 1;
    const float wy0_l = (y0i_l >= 0 && y0i_l < HH) ? (1.0f - fy_l) : 0.0f;
    const float wy1_l = (y1i_l >= 0 && y1i_l < HH) ? fy_l : 0.0f;
    const int   y0c_l = iclamp(y0i_l, 0, HH - 1);
    const int   y1c_l = iclamp(y1i_l, 0, HH - 1);

    // ---- Build A: y-tile as 16x32 A operand, contiguous k order (same as B) ----
    // lane L (m = L&15): VGPR0..7 = K kba..kba+15 linear,
    // kba = (L<16) ? 0 : 16.
    const int kba = (lane < 16) ? 0 : 16;
    const float* yrow = y + (((size_t)b * HH + h) * WW + pw) * CC;
    f32x4 ya0 = *(const f32x4*)(yrow + kba + 0);
    f32x4 ya1 = *(const f32x4*)(yrow + kba + 4);
    f32x4 ya2 = *(const f32x4*)(yrow + kba + 8);
    f32x4 ya3 = *(const f32x4*)(yrow + kba + 12);
    v16h A;
    A[0]  = (_Float16)ya0.x; A[1]  = (_Float16)ya0.y;
    A[2]  = (_Float16)ya0.z; A[3]  = (_Float16)ya0.w;
    A[4]  = (_Float16)ya1.x; A[5]  = (_Float16)ya1.y;
    A[6]  = (_Float16)ya1.z; A[7]  = (_Float16)ya1.w;
    A[8]  = (_Float16)ya2.x; A[9]  = (_Float16)ya2.y;
    A[10] = (_Float16)ya2.z; A[11] = (_Float16)ya2.w;
    A[12] = (_Float16)ya3.x; A[13] = (_Float16)ya3.y;
    A[14] = (_Float16)ya3.z; A[15] = (_Float16)ya3.w;

    // B layout: lane L holds column n=L&15, channels kbb..kbb+15 linear,
    // kbb = (L<16) ? 0 : 16.
    const int kbb = (lane < 16) ? 0 : 16;
    const float* xb = x + (size_t)b * HH * WW * CC;

    // Output base for the diag-owning lanes: lanes 0..7 -> j=lane,
    // lanes 24..31 -> j=lane-16.
    const bool diag_lane = (lane < 8) || (lane >= 24);
    const int  j         = (lane < 8) ? lane : (lane - 16);
    float* outp = out + (((size_t)b * HH + h) * WW + (w0 + j)) * SS;

    for (int d = 0; d < SS; ++d) {
        // Broadcast this step's parameters from lane d (uniform d -> SGPR).
        const float alpha = bcast_f(alpha_l, d);
        const float beta  = bcast_f(beta_l, d);
        const float wy0   = bcast_f(wy0_l, d);
        const float wy1   = bcast_f(wy1_l, d);
        const int   y0c   = bcast_i(y0c_l, d);
        const int   y1c   = bcast_i(y1c_l, d);

        // Column sampling (per pixel).
        const float sx  = alpha * (float)pw + beta;
        const float x0f = floorf(sx);
        const float fxr = sx - x0f;
        const int   x0i = (int)x0f;
        const int   x1i = x0i + 1;
        const float wx0 = (x0i >= 0 && x0i < WW) ? (1.0f - fxr) : 0.0f;
        const float wx1 = (x1i >= 0 && x1i < WW) ? fxr : 0.0f;
        const int   x0c = iclamp(x0i, 0, WW - 1);
        const int   x1c = iclamp(x1i, 0, WW - 1);

        const float w00 = wy0 * wx0, w01 = wy0 * wx1;
        const float w10 = wy1 * wx0, w11 = wy1 * wx1;

        const float* p00 = xb + ((size_t)y0c * WW + x0c) * CC + kbb;
        const float* p01 = xb + ((size_t)y0c * WW + x1c) * CC + kbb;
        const float* p10 = xb + ((size_t)y1c * WW + x0c) * CC + kbb;
        const float* p11 = xb + ((size_t)y1c * WW + x1c) * CC + kbb;

        // ---- Build B: 16 interpolated channels of this lane's pixel ----
        v16h Bv;
#pragma unroll
        for (int q = 0; q < 4; ++q) {
            f32x4 a0 = *(const f32x4*)(p00 + 4 * q);
            f32x4 a1 = *(const f32x4*)(p01 + 4 * q);
            f32x4 a2 = *(const f32x4*)(p10 + 4 * q);
            f32x4 a3 = *(const f32x4*)(p11 + 4 * q);
            f32x4 xw = w00 * a0 + w01 * a1 + w10 * a2 + w11 * a3;
            Bv[4 * q + 0] = (_Float16)xw.x;
            Bv[4 * q + 1] = (_Float16)xw.y;
            Bv[4 * q + 2] = (_Float16)xw.z;
            Bv[4 * q + 3] = (_Float16)xw.w;
        }

        // ---- WMMA: C[i,j] = sum_c y[w_i,c] * warpx[w_j,c] ----
        v8f Cz = {};
        v8f Cd = __builtin_amdgcn_wmma_f32_16x16x32_f16(
            /*neg_a=*/false, A, /*neg_b=*/false, Bv,
            /*c_mod=*/(short)0, Cz, /*reuse_a=*/false, /*reuse_b=*/false);

        // ---- Extract diagonal: lane L<8 has diag L in Cd[L];
        //      lane L>=24 has diag (L-16) in Cd[L-24]. Select r = lane&7. ----
        const int r = lane & 7;
        float dv = 0.0f;
#pragma unroll
        for (int rr = 0; rr < 8; ++rr)
            if (r == rr) dv = Cd[rr];

        if (diag_lane)
            outp[d] = dv * (1.0f / 32.0f);
    }
}

extern "C" void kernel_launch(void* const* d_in, const int* in_sizes, int n_in,
                              void* d_out, int out_size, void* d_ws, size_t ws_size,
                              hipStream_t stream) {
    const float* x      = (const float*)d_in[0];
    const float* y      = (const float*)d_in[1];
    const float* origin = (const float*)d_in[2];
    const float* focal  = (const float*)d_in[3];
    const float* T12    = (const float*)d_in[4];
    float* out          = (float*)d_out;

    // 128 pixels per 256-thread block (8 waves x 16 px); W=640 tiles exactly.
    const int grid = BB * HH * (WW / 128);
    corr_wmma_kernel<<<grid, 256, 0, stream>>>(x, y, origin, focal, T12, out);
}
